// Aggregator_10720238371091
// MI455X (gfx1250) — matrix operand order repaired and checked by reading, not compiled, not run on hardware
//
#include <hip/hip_runtime.h>

#define IN_DIM 128
#define OUT_DIM 128
#define NSEG 16384
#define EPS 1e-5f
#define LDSW_STRIDE  136  // 128 halves + 8 pad -> 272B row stride (4-bank rot/row)
#define LDSXF_STRIDE 132  // 128 floats + 4 pad -> 528B row stride (4-bank rot/row)
#define LDSXB_STRIDE 136  // x as bf16: 128 halves + 8 pad
#define LDSXF_BYTES (128 * LDSXF_STRIDE * 4)             // 67,584
#define LDSXB_BYTES (128 * LDSXB_STRIDE * 2)             // 34,816
#define LDSW_BYTES  (OUT_DIM * LDSW_STRIDE * 2)          // 34,816
#define LDS_BYTES   (LDSXF_BYTES + LDSXB_BYTES + LDSW_BYTES)  // 137,216

typedef __attribute__((ext_vector_type(16))) __bf16 v16bf;
typedef __attribute__((ext_vector_type(2)))  __bf16 v2bf;
typedef __attribute__((ext_vector_type(8)))  float  v8f;
typedef __attribute__((ext_vector_type(4)))  int    v4i;

// async-copy builtin takes pointers to 4xi32 vectors in explicit addr spaces
typedef __attribute__((address_space(1))) v4i* gas_b128_t;
typedef __attribute__((address_space(3))) v4i* las_b128_t;

#ifndef __has_builtin
#define __has_builtin(x) 0
#endif

#if __has_builtin(__builtin_amdgcn_global_load_async_to_lds_b128) && \
    __has_builtin(__builtin_amdgcn_s_wait_asynccnt)
#define HAVE_ASYNC_COPY 1
#else
#define HAVE_ASYNC_COPY 0
#endif

#if __has_builtin(__builtin_amdgcn_cvt_pk_bf16_f32)
#define HAVE_CVT_PK 1
#else
#define HAVE_CVT_PK 0
#endif

union FragBF {
    v16bf v;
    uint4 q[2];
};

// two floats -> packed bf16x2.
// Fast path: hardware packed convert if available. Fallback: round-to-nearest
// via +0x8000 then a single v_perm_b32 extracting both high halves (3 VALU).
__device__ __forceinline__ unsigned int pack2bf(float a, float b) {
#if HAVE_CVT_PK
    v2bf p = __builtin_amdgcn_cvt_pk_bf16_f32(a, b);
    return __builtin_bit_cast(unsigned int, p);
#else
    unsigned int au = __float_as_uint(a) + 0x8000u;
    unsigned int bu = __float_as_uint(b) + 0x8000u;
    // bytes {bu[3],bu[2],au[3],au[2]} = {b_hi16, a_hi16}
    return __builtin_amdgcn_perm(bu, au, 0x07060302u);
#endif
}

// ---------------------------------------------------------------------------
// Kernel 1: zero segment sums (d_out) and counts. NSEG*OUT_DIM threads.
__global__ __launch_bounds__(256) void zero_kernel(float* __restrict__ out,
                                                   float* __restrict__ cnt) {
    int i = blockIdx.x * 256 + threadIdx.x;
    out[i] = 0.0f;
    if (i < NSEG) cnt[i] = 0.0f;
}

// ---------------------------------------------------------------------------
// Kernel 2: per-segment counts.
__global__ __launch_bounds__(256) void count_kernel(const int* __restrict__ batch,
                                                    float* __restrict__ cnt, int n) {
    int i = blockIdx.x * 256 + threadIdx.x;
    if (i < n) atomicAdd(cnt + batch[i], 1.0f);
}

// ---------------------------------------------------------------------------
// Kernel 3: fused  h = x@W^T + b -> LayerNorm -> atomic segment-sum.
// Block = 256 threads = 8 waves; block tile = 128 rows.
//   stage 1: async-copy raw f32 x tile (64KB) to LDS; convert W to bf16 in LDS
//   stage 2: one pass converts the x tile to bf16 in a second LDS region
//   stage 3: per wave, 4 K-steps x 8 N-tiles of pure ds_load + v_wmma
//   stage 4: in-register LayerNorm (width-16 shfl reductions), atomic scatter
__global__ __launch_bounds__(256) void fused_gemm_ln_scatter(
        const float* __restrict__ x, const int* __restrict__ batch,
        const float* __restrict__ W, const float* __restrict__ bvec,
        const float* __restrict__ lnw, const float* __restrict__ lnb,
        float* __restrict__ segsum, int n) {
    extern __shared__ __align__(16) char smem[];
    float* ldsXf = (float*)smem;                                      // 128x132 f32
    unsigned short* ldsXb = (unsigned short*)(smem + LDSXF_BYTES);    // 128x136 bf16
    unsigned short* ldsW  = (unsigned short*)(smem + LDSXF_BYTES + LDSXB_BYTES);

    const long long rowBase0 = (long long)blockIdx.x * 128;

    // ---- stage W as bf16 (packed stores: 2 halves per uint) ----
    for (int idx = threadIdx.x; idx < (OUT_DIM * IN_DIM) / 2; idx += 256) {
        int o = idx >> 6, i2 = (idx & 63) * 2;
        const float* wp = W + o * IN_DIM + i2;
        *(unsigned int*)(ldsW + o * LDSW_STRIDE + i2) = pack2bf(wp[0], wp[1]);
    }

    // ---- stage x tile: 128 rows x 512B = 4096 x b128 transfers ----
#if HAVE_ASYNC_COPY
    for (int t = threadIdx.x; t < 4096; t += 256) {
        int r = t >> 5, c = t & 31;                 // one wave = one full row
        long long gr = rowBase0 + r;
        if (gr >= n) gr = n - 1;
        const float* gp = x + gr * IN_DIM + c * 4;
        float* lp = ldsXf + r * LDSXF_STRIDE + c * 4;
        __builtin_amdgcn_global_load_async_to_lds_b128(
            (gas_b128_t)(const void*)gp, (las_b128_t)(void*)lp, 0, 0);
    }
    __builtin_amdgcn_s_wait_asynccnt(0);
#else
    for (int t = threadIdx.x; t < 4096; t += 256) {
        int r = t >> 5, c = t & 31;
        long long gr = rowBase0 + r;
        if (gr >= n) gr = n - 1;
        *(float4*)(ldsXf + r * LDSXF_STRIDE + c * 4) =
            *(const float4*)(x + gr * IN_DIM + c * 4);
    }
#endif
    __syncthreads();

    // ---- convert x tile to bf16 once (8192 float pairs) ----
    for (int p = threadIdx.x; p < (128 * IN_DIM) / 2; p += 256) {
        int r = p >> 6, c2 = (p & 63) * 2;
        const float* sp = ldsXf + r * LDSXF_STRIDE + c2;
        *(unsigned int*)(ldsXb + r * LDSXB_STRIDE + c2) = pack2bf(sp[0], sp[1]);
    }
    __syncthreads();

    const int wave = threadIdx.x >> 5;
    const int lane = threadIdx.x & 31;
    const int g    = lane >> 4;      // half-wave group
    const int ln   = lane & 15;      // N index / lane-in-group
    const long long rowBase = rowBase0 + wave * 16;

    // this lane's bf16 A row (272B row stride -> conflict-free across lanes)
    const unsigned short* xbr = ldsXb + (wave * 16 + ln) * LDSXB_STRIDE;

    v8f acc[8];
#pragma unroll
    for (int ct = 0; ct < 8; ++ct) acc[ct] = (v8f){0, 0, 0, 0, 0, 0, 0, 0};

#pragma unroll
    for (int kt = 0; kt < 4; ++kt) {
        // A fragment (16x32 bf16): halves 0..7 = K kt*32+g*8.., halves 8..15 =
        // K kt*32+16+g*8.. (ISA 16-bit A layout) -> two 16B LDS reads.
        FragBF a;
        a.q[0] = *(const uint4*)(xbr + kt * 32 + g * 8);
        a.q[1] = *(const uint4*)(xbr + kt * 32 + 16 + g * 8);

#pragma unroll
        for (int ct = 0; ct < 8; ++ct) {
            // B fragment (32x16 bf16): W row o = ct*16+ln, halves 0..7 = K
            // kt*32+g*8.., halves 8..15 = K kt*32+16+g*8.. (same as A)
            FragBF bf;
            const uint4* p = (const uint4*)(ldsW + (ct * 16 + ln) * LDSW_STRIDE
                                            + kt * 32 + g * 8);
            bf.q[0] = p[0];
            bf.q[1] = p[2];
            acc[ct] = __builtin_amdgcn_wmma_f32_16x16x32_bf16(
                false, a.v, false, bf.v, (short)0, acc[ct], false, false);
        }
    }

    // ---- bias + LayerNorm stats (row m = g*8 + v lives in acc[*][v]) ----
    float biasv[8], lnwv[8], lnbv[8];
#pragma unroll
    for (int ct = 0; ct < 8; ++ct) {
        int col = ct * 16 + ln;
        biasv[ct] = bvec[col];
        lnwv[ct]  = lnw[col];
        lnbv[ct]  = lnb[col];
    }
    float s1[8], s2[8];
#pragma unroll
    for (int v = 0; v < 8; ++v) { s1[v] = 0.0f; s2[v] = 0.0f; }
#pragma unroll
    for (int ct = 0; ct < 8; ++ct) {
#pragma unroll
        for (int v = 0; v < 8; ++v) {
            float t = acc[ct][v] + biasv[ct];
            acc[ct][v] = t;
            s1[v] += t;
            s2[v] += t * t;
        }
    }
#pragma unroll
    for (int m = 8; m >= 1; m >>= 1) {
#pragma unroll
        for (int v = 0; v < 8; ++v) {
            s1[v] += __shfl_xor(s1[v], m, 16);
            s2[v] += __shfl_xor(s2[v], m, 16);
        }
    }
    float mean[8], rstd[8];
#pragma unroll
    for (int v = 0; v < 8; ++v) {
        float mu  = s1[v] * (1.0f / 128.0f);
        float var = s2[v] * (1.0f / 128.0f) - mu * mu;
        mean[v] = mu;
        rstd[v] = rsqrtf(var + EPS);
    }

    // ---- normalize + atomic scatter into segment sums (L2-resident) ----
    int seg[8];
    const long long rbase = rowBase + g * 8;
#pragma unroll
    for (int v = 0; v < 8; ++v) {
        long long r = rbase + v;
        seg[v] = (r < n) ? batch[r] : -1;
    }
#pragma unroll
    for (int ct = 0; ct < 8; ++ct) {
#pragma unroll
        for (int v = 0; v < 8; ++v) {
            if (seg[v] >= 0) {
                float y = (acc[ct][v] - mean[v]) * rstd[v] * lnwv[ct] + lnbv[ct];
                atomicAdd(segsum + (long long)seg[v] * OUT_DIM + ct * 16 + ln, y);
            }
        }
    }
}

// ---------------------------------------------------------------------------
// Kernel 4: divide segment sums by clamped counts. NSEG*OUT_DIM threads.
__global__ __launch_bounds__(256) void finalize_kernel(float* __restrict__ out,
                                                       const float* __restrict__ cnt) {
    int i = blockIdx.x * 256 + threadIdx.x;
    float c = cnt[i >> 7];
    out[i] = out[i] / fmaxf(c, 1.0f);
}

extern "C" void kernel_launch(void* const* d_in, const int* in_sizes, int n_in,
                              void* d_out, int out_size, void* d_ws, size_t ws_size,
                              hipStream_t stream) {
    const float* x     = (const float*)d_in[0];
    const int*   batch = (const int*)d_in[1];
    const float* W     = (const float*)d_in[2];
    const float* bvec  = (const float*)d_in[3];
    const float* lnw   = (const float*)d_in[4];
    const float* lnb   = (const float*)d_in[5];
    float* out = (float*)d_out;
    float* cnt = (float*)d_ws;   // NSEG floats of scratch

    const int n = in_sizes[0] / IN_DIM;

    // NSEG*OUT_DIM = 2,097,152 = 8192 * 256
    zero_kernel<<<8192, 256, 0, stream>>>(out, cnt);
    if (n > 0) {
        count_kernel<<<(n + 255) / 256, 256, 0, stream>>>(batch, cnt, n);
        const int nblk = (n + 127) / 128;   // 128 rows per block (8 waves x 16)
        fused_gemm_ln_scatter<<<nblk, 256, LDS_BYTES, stream>>>(
            x, batch, W, bvec, lnw, lnb, out, n);
    }
    finalize_kernel<<<8192, 256, 0, stream>>>(out, cnt);
}
